// PalmBridge_20109036880003
// MI455X (gfx1250) — matrix operand order repaired and checked by reading, not compiled, not run on hardware
//
#include <hip/hip_runtime.h>
#include <hip/hip_bf16.h>

typedef __attribute__((ext_vector_type(16))) __bf16          v16bf;
typedef __attribute__((ext_vector_type(16))) unsigned short  v16u;
typedef __attribute__((ext_vector_type(8)))  float           v8f;

#define N_ROWS          131072
#define K_CODES         512
#define DDIM            512
#define ROWS_PER_BLOCK  64
#define BLOCK_THREADS   128
#define W_ORI           0.7f
#define W_MAP           0.3f

// round-to-nearest-even f32 -> bf16 (bit pattern)
__device__ __forceinline__ unsigned short f2bf(float f) {
    union { float f; unsigned u; } x; x.f = f;
    unsigned u = x.u + 0x7FFFu + ((x.u >> 16) & 1u);
    return (unsigned short)(u >> 16);
}

// ---------------------------------------------------------------------------
// Prep 1: build WMMA-ready bf16 B image of P^T in ws.
// Layout: block b = ct*16 + t (code-tile ct, k-tile t), then lane l (0..31),
// then position n (0..15, the k offset within the lane's 16-k half).
//   Bw[((ct*16+t)*32 + l)*16 + n] = bf16( P[(ct*16+(l&15))*D + t*32 + (l>>4)*16 + n] )
// contiguous K order per lane half, matching the A-tile load.
// ---------------------------------------------------------------------------
__global__ void vq_bprep(const float* __restrict__ P,
                         unsigned short* __restrict__ Bw) {
    int gid = blockIdx.x * 256 + threadIdx.x;      // 0 .. 512*512-1
    int n   = gid & 15;
    int l   = (gid >> 4) & 31;
    int b   = gid >> 9;
    int t   = b & 15;
    int ct  = b >> 4;
    int d   = t * 32 + (l >> 4) * 16 + n;
    int code = ct * 16 + (l & 15);
    Bw[gid] = f2bf(P[code * DDIM + d]);
}

// ---------------------------------------------------------------------------
// Prep 2: pnorm[k] = ||P[k]||^2 in f32
// ---------------------------------------------------------------------------
__global__ void vq_pnorm(const float* __restrict__ P, float* __restrict__ pn) {
    int k = blockIdx.x * 256 + threadIdx.x;
    if (k < K_CODES) {
        float s = 0.f;
        const float* row = P + k * DDIM;
        #pragma unroll 8
        for (int d = 0; d < DDIM; ++d) { float v = row[d]; s += v * v; }
        pn[k] = s;
    }
}

// ---------------------------------------------------------------------------
// Main fused kernel: 128 threads = 4 waves, 64 rows per block (16 per wave).
// ---------------------------------------------------------------------------
__global__ __launch_bounds__(BLOCK_THREADS, 1)
void vq_main(const float* __restrict__ z,
             const float* __restrict__ P,
             const unsigned short* __restrict__ Bw,
             const float* __restrict__ pn,
             float* __restrict__ zhat,
             float* __restrict__ ztil,
             float* __restrict__ idxOut) {
    extern __shared__ float smem[];
    float*          zs = smem;                                   // 64*512 f32 = 128 KB
    unsigned short* bs = (unsigned short*)(smem + ROWS_PER_BLOCK * DDIM); // 16 KB

    const int tid  = threadIdx.x;
    const int lane = tid & 31;
    const int w    = tid >> 5;                  // wave id 0..3
    const long long rowBase = (long long)blockIdx.x * ROWS_PER_BLOCK;

    // ---- phase 1: stage the 64x512 f32 z tile into LDS (coalesced) ----
    {
        const float4* zg = (const float4*)(z + rowBase * DDIM);
        float4*       zl = (float4*)zs;
        #pragma unroll 8
        for (int i = 0; i < 64; ++i)            // 8192 float4 / 128 threads
            zl[i * BLOCK_THREADS + tid] = zg[i * BLOCK_THREADS + tid];
    }
    __syncthreads();

    // ---- phase 2: build bf16 A tiles in registers (16 rows x 512 k) ----
    const int m    = lane & 15;
    const int half = lane >> 4;
    const float* zrow = zs + (w * 16 + m) * DDIM;
    union AU { v16u u; v16bf b; };
    AU a[16];
    #pragma unroll
    for (int t = 0; t < 16; ++t) {
        const float* src = zrow + t * 32 + half * 16;
        #pragma unroll
        for (int j = 0; j < 16; ++j) a[t].u[j] = f2bf(src[j]);
    }

    // ---- phase 3: GEMM over 32 code tiles, track per-lane argmin ----
    float best[8];
    int   bidx[8];
    #pragma unroll
    for (int r = 0; r < 8; ++r) { best[r] = 3.4e38f; bidx[r] = 0; }

    for (int ct = 0; ct < 32; ++ct) {
        __syncthreads();                        // protect previous B tile
        {   // cooperative B stage: 16 KB = 1024 uint4, 8 per thread
            const uint4* src = (const uint4*)(Bw + ct * 8192);
            uint4*       dst = (uint4*)bs;
            #pragma unroll
            for (int i = 0; i < 8; ++i)
                dst[i * BLOCK_THREADS + tid] = src[i * BLOCK_THREADS + tid];
        }
        __syncthreads();

        v8f acc = {};
        #pragma unroll
        for (int t = 0; t < 16; ++t) {
            v16bf bb = *(const v16bf*)(bs + t * 512 + lane * 16);
            acc = __builtin_amdgcn_wmma_f32_16x16x32_bf16(
                false, a[t].b, false, bb, (short)0, acc, false, false);
        }

        int   code = ct * 16 + (lane & 15);
        float pnv  = pn[code];
        #pragma unroll
        for (int r = 0; r < 8; ++r) {
            float sc = pnv - 2.0f * acc[r];     // ||p||^2 - 2 z.p
            if (sc < best[r]) { best[r] = sc; bidx[r] = code; }  // strict < keeps lowest idx
        }
    }

    // ---- phase 4: cross-lane argmin within each 16-lane half ----
    #pragma unroll
    for (int r = 0; r < 8; ++r) {
        #pragma unroll
        for (int msk = 1; msk <= 8; msk <<= 1) {
            float so = __shfl_xor(best[r], msk, 32);
            int   io = __shfl_xor(bidx[r], msk, 32);
            if (so < best[r] || (so == best[r] && io < bidx[r])) {
                best[r] = so; bidx[r] = io;
            }
        }
    }
    // lane-local select: v = bidx[lane & 7]
    int v = bidx[0];
    #pragma unroll
    for (int r = 1; r < 8; ++r) v = ((lane & 7) == r) ? bidx[r] : v;
    // row (lane & 15)'s best index lives on lane ((m>>3)<<4) | (m&7)
    int rowIdx = __shfl(v, (((lane >> 3) & 1) << 4) | (lane & 7), 32);
    if (lane < 16)
        idxOut[rowBase + w * 16 + lane] = (float)rowIdx;

    // ---- phase 5: gather P[idx], blend, store (coalesced per row) ----
    #pragma unroll
    for (int mm = 0; mm < 16; ++mm) {
        int im = __shfl(v, ((mm >> 3) << 4) | (mm & 7), 32);
        const float4* prow = (const float4*)(P + (long long)im * DDIM);
        const float4* zl   = (const float4*)(zs + (w * 16 + mm) * DDIM);
        long long gr = (rowBase + w * 16 + mm) * DDIM;
        float4* zh = (float4*)(zhat + gr);
        float4* zt = (float4*)(ztil + gr);
        #pragma unroll
        for (int j = 0; j < 4; ++j) {
            int o = j * 32 + lane;              // 128 float4 per row
            float4 p4 = prow[o];
            float4 z4 = zl[o];
            float4 h;
            h.x = W_ORI * z4.x + W_MAP * p4.x;
            h.y = W_ORI * z4.y + W_MAP * p4.y;
            h.z = W_ORI * z4.z + W_MAP * p4.z;
            h.w = W_ORI * z4.w + W_MAP * p4.w;
            zt[o] = p4;
            zh[o] = h;
        }
    }
}

// ---------------------------------------------------------------------------
extern "C" void kernel_launch(void* const* d_in, const int* in_sizes, int n_in,
                              void* d_out, int out_size, void* d_ws, size_t ws_size,
                              hipStream_t stream) {
    const float* z = (const float*)d_in[0];
    const float* P = (const float*)d_in[1];

    unsigned short* Bw = (unsigned short*)d_ws;                    // 512 KB
    float*          pn = (float*)((char*)d_ws + (size_t)512 * 1024); // 2 KB

    float* out  = (float*)d_out;
    float* zhat = out;
    float* ztil = out + (long long)N_ROWS * DDIM;
    float* idxO = out + 2LL * N_ROWS * DDIM;

    vq_bprep<<<(K_CODES * DDIM) / 256, 256, 0, stream>>>(P, Bw);
    vq_pnorm<<<(K_CODES + 255) / 256, 256, 0, stream>>>(P, pn);

    size_t shmem = (size_t)ROWS_PER_BLOCK * DDIM * sizeof(float) + 16 * 1024;
    vq_main<<<N_ROWS / ROWS_PER_BLOCK, BLOCK_THREADS, shmem, stream>>>(
        z, P, Bw, pn, zhat, ztil, idxO);
}
